// DomainAdaptationModule_42992622633718
// MI455X (gfx1250) — matrix operand order repaired and checked by reading, not compiled, not run on hardware
//
#include <hip/hip_runtime.h>
#include <hip/hip_bf16.h>
#include <stdint.h>

#define B_   64
#define L_   512
#define D_   768
#define E_   10
#define LN_EPS 1e-5f

typedef __attribute__((ext_vector_type(16))) __bf16 v16bf;
typedef __attribute__((ext_vector_type(8)))  float  v8f;

// ---------------------------------------------------------------------------
// Kernel 1: one-time f32 -> bf16 conversion of the expert weights into d_ws.
// W is (E, D, D) row-major; bf16 copy keeps the same layout.
// ---------------------------------------------------------------------------
__global__ __launch_bounds__(256)
void cvt_w_to_bf16(const float* __restrict__ in, __bf16* __restrict__ out, int n) {
    int i = (blockIdx.x * blockDim.x + threadIdx.x) * 8;
    if (i + 8 > n) return;
    float4 a = *(const float4*)(in + i);
    float4 c = *(const float4*)(in + i + 4);
    union { __bf16 h[8]; uint4 u; } pk;
    pk.h[0] = (__bf16)a.x; pk.h[1] = (__bf16)a.y;
    pk.h[2] = (__bf16)a.z; pk.h[3] = (__bf16)a.w;
    pk.h[4] = (__bf16)c.x; pk.h[5] = (__bf16)c.y;
    pk.h[6] = (__bf16)c.z; pk.h[7] = (__bf16)c.w;
    *(uint4*)(out + i) = pk.u;
}

// ---------------------------------------------------------------------------
// Kernel 2: per-sample expert GEMM (bf16 WMMA, f32 accum) + fused LayerNorm.
// Block: 256 threads = 8 waves; block computes 32 L-rows x 768 O-cols.
// Wave tile: 32 (M) x 96 (N)  ->  2 A fragments x 6 B fragments, 12 accums.
// Each B fragment feeds 2 WMMAs (register-level reuse, halves W traffic).
// ---------------------------------------------------------------------------
__global__ __launch_bounds__(256)
void moe_gemm_ln(const float*  __restrict__ X,      // (B, L, D) f32
                 const __bf16* __restrict__ Wb,     // (E, D, D) bf16
                 const float*  __restrict__ bias,   // (E, D)
                 const float*  __restrict__ gamma,  // (E, D)
                 const float*  __restrict__ beta,   // (E, D)
                 const int*    __restrict__ ids,    // (B)
                 float*        __restrict__ out)    // (B, L, D)
{
    const int bidL  = blockIdx.x;          // 0..15  (L tile)
    const int b     = blockIdx.y;          // 0..63  (sample)
    const int e     = ids[b];              // expert id

    const int tid   = threadIdx.x;
    const int wave  = tid >> 5;            // 0..7 -> which 96-column slab
    const int lane  = tid & 31;
    const int lhalf = lane & 15;
    const int sel   = lane >> 4;           // 0 (lanes 0-15) / 1 (lanes 16-31)

    const int lbase = bidL * 32;           // base L row of this block
    const int nbase = wave * 96;           // base output column of this wave

    __shared__ float s_sum[32][8];
    __shared__ float s_sq [32][8];
    __shared__ float s_mean[32];
    __shared__ float s_rstd[32];

    v8f acc[2][6];
    #pragma unroll
    for (int mt = 0; mt < 2; ++mt)
        #pragma unroll
        for (int j = 0; j < 6; ++j)
            acc[mt][j] = (v8f){0.f,0.f,0.f,0.f,0.f,0.f,0.f,0.f};

    // A-fragment row bases for this lane (16-bit A 16x32 layout:
    //  lanes 0-15 : K chunks at +0..7 and +16..23
    //  lanes 16-31: K chunks at +8..15 and +24..31)  -> fold sel*8 into base.
    const float*  Xr0 = X + ((size_t)b * L_ + (lbase + lhalf)) * D_ + sel * 8;
    const float*  Xr1 = Xr0 + (size_t)16 * D_;
    // B base: column (nbase + lhalf); per-lane K chunks at k0 + sel*8 + 0..7 and k0 + 16 + sel*8 + 0..7.
    // The 6 N-tiles sit at compile-time offsets j*16*D (24KB) -> ioffset imm.
    const __bf16* Wbase = Wb + (size_t)e * D_ * D_
                             + (size_t)(nbase + lhalf) * D_ + sel * 8;

    for (int k0 = 0; k0 < D_; k0 += 32) {
        // ---- issue all 6 B fragments first (distinct buffers, clause-able)
        union { v16bf v; uint4 u[2]; } bf[6];
        #pragma unroll
        for (int j = 0; j < 6; ++j) {
            const __bf16* pb = Wbase + k0 + j * (16 * D_);
            bf[j].u[0] = *(const uint4*)(pb + 0);
            bf[j].u[1] = *(const uint4*)(pb + 16);
        }

        // ---- A fragments for the two M-tiles (f32 load + pack to bf16)
        union { v16bf v; __bf16 h[16]; } a0, a1;
        {
            const float* pa = Xr0 + k0;
            float4 x0 = *(const float4*)(pa + 0);
            float4 x1 = *(const float4*)(pa + 4);
            float4 x2 = *(const float4*)(pa + 16);
            float4 x3 = *(const float4*)(pa + 20);
            a0.h[0] = (__bf16)x0.x; a0.h[1] = (__bf16)x0.y; a0.h[2]  = (__bf16)x0.z; a0.h[3]  = (__bf16)x0.w;
            a0.h[4] = (__bf16)x1.x; a0.h[5] = (__bf16)x1.y; a0.h[6]  = (__bf16)x1.z; a0.h[7]  = (__bf16)x1.w;
            a0.h[8] = (__bf16)x2.x; a0.h[9] = (__bf16)x2.y; a0.h[10] = (__bf16)x2.z; a0.h[11] = (__bf16)x2.w;
            a0.h[12]= (__bf16)x3.x; a0.h[13]= (__bf16)x3.y; a0.h[14] = (__bf16)x3.z; a0.h[15] = (__bf16)x3.w;
            __builtin_prefetch(pa + 32, 0, 1);     // next K-step A row (global_prefetch)
        }
        {
            const float* pa = Xr1 + k0;
            float4 x0 = *(const float4*)(pa + 0);
            float4 x1 = *(const float4*)(pa + 4);
            float4 x2 = *(const float4*)(pa + 16);
            float4 x3 = *(const float4*)(pa + 20);
            a1.h[0] = (__bf16)x0.x; a1.h[1] = (__bf16)x0.y; a1.h[2]  = (__bf16)x0.z; a1.h[3]  = (__bf16)x0.w;
            a1.h[4] = (__bf16)x1.x; a1.h[5] = (__bf16)x1.y; a1.h[6]  = (__bf16)x1.z; a1.h[7]  = (__bf16)x1.w;
            a1.h[8] = (__bf16)x2.x; a1.h[9] = (__bf16)x2.y; a1.h[10] = (__bf16)x2.z; a1.h[11] = (__bf16)x2.w;
            a1.h[12]= (__bf16)x3.x; a1.h[13]= (__bf16)x3.y; a1.h[14] = (__bf16)x3.z; a1.h[15] = (__bf16)x3.w;
            __builtin_prefetch(pa + 32, 0, 1);
        }

        // ---- 12 WMMAs; each B fragment reused for both M-tiles
        #pragma unroll
        for (int j = 0; j < 6; ++j) {
            acc[0][j] = __builtin_amdgcn_wmma_f32_16x16x32_bf16(
                            false, a0.v, false, bf[j].v, (short)0, acc[0][j], false, false);
            acc[1][j] = __builtin_amdgcn_wmma_f32_16x16x32_bf16(
                            false, a1.v, false, bf[j].v, (short)0, acc[1][j], false, false);
        }
    }

    // ------------------- epilogue: bias + LayerNorm partials -------------------
    // C layout: VGPR r, lane -> (M = r + sel*8, N = lhalf)
    float s[2][8], q[2][8];
    #pragma unroll
    for (int mt = 0; mt < 2; ++mt)
        #pragma unroll
        for (int r = 0; r < 8; ++r) { s[mt][r] = 0.f; q[mt][r] = 0.f; }

    #pragma unroll
    for (int j = 0; j < 6; ++j) {
        const float bj = bias[(size_t)e * D_ + nbase + j * 16 + lhalf];
        #pragma unroll
        for (int mt = 0; mt < 2; ++mt)
            #pragma unroll
            for (int r = 0; r < 8; ++r) {
                float v = acc[mt][j][r] + bj;
                acc[mt][j][r] = v;
                s[mt][r] += v;
                q[mt][r] += v * v;
            }
    }

    // reduce across each 16-lane half (a half shares the same set of rows)
    #pragma unroll
    for (int mt = 0; mt < 2; ++mt)
        #pragma unroll
        for (int r = 0; r < 8; ++r) {
            #pragma unroll
            for (int m = 1; m < 16; m <<= 1) {
                s[mt][r] += __shfl_xor(s[mt][r], m, 32);
                q[mt][r] += __shfl_xor(q[mt][r], m, 32);
            }
        }
    if (lhalf == 0) {   // lanes 0 and 16 hold the half-wave sums
        #pragma unroll
        for (int mt = 0; mt < 2; ++mt)
            #pragma unroll
            for (int r = 0; r < 8; ++r) {
                s_sum[mt * 16 + r + sel * 8][wave] = s[mt][r];
                s_sq [mt * 16 + r + sel * 8][wave] = q[mt][r];
            }
    }
    __syncthreads();

    if (tid < 32) {     // one thread per L-row of this block
        float su = 0.f, sq = 0.f;
        #pragma unroll
        for (int g = 0; g < 8; ++g) { su += s_sum[tid][g]; sq += s_sq[tid][g]; }
        float mean = su * (1.0f / (float)D_);
        float var  = sq * (1.0f / (float)D_) - mean * mean;
        s_mean[tid] = mean;
        s_rstd[tid] = rsqrtf(var + LN_EPS);
    }
    __syncthreads();

    // ------------------- apply LN affine and store -------------------
    #pragma unroll
    for (int j = 0; j < 6; ++j) {
        const int o  = nbase + j * 16 + lhalf;
        const float gm = gamma[(size_t)e * D_ + o];
        const float bt = beta [(size_t)e * D_ + o];
        #pragma unroll
        for (int mt = 0; mt < 2; ++mt)
            #pragma unroll
            for (int r = 0; r < 8; ++r) {
                const int lrow = mt * 16 + r + sel * 8;
                const float v  = (acc[mt][j][r] - s_mean[lrow]) * s_rstd[lrow] * gm + bt;
                out[((size_t)b * L_ + lbase + lrow) * D_ + o] = v;
            }
    }
}

// ---------------------------------------------------------------------------
// Host launcher
// ---------------------------------------------------------------------------
extern "C" void kernel_launch(void* const* d_in, const int* in_sizes, int n_in,
                              void* d_out, int out_size, void* d_ws, size_t ws_size,
                              hipStream_t stream) {
    const float* X     = (const float*)d_in[0];   // embeddings (B,L,D)
    const float* W     = (const float*)d_in[1];   // W (E,D,D)
    const float* bias  = (const float*)d_in[2];   // b (E,D)
    const float* gamma = (const float*)d_in[3];   // gamma (E,D)
    const float* beta  = (const float*)d_in[4];   // beta (E,D)
    const int*   ids   = (const int*)d_in[5];     // domain_ids (B)
    float*       out   = (float*)d_out;

    __bf16* Wbf = (__bf16*)d_ws;                  // needs E*D*D*2 = ~11.25 MB
    const int nW = E_ * D_ * D_;

    cvt_w_to_bf16<<<dim3(nW / (256 * 8)), dim3(256), 0, stream>>>(W, Wbf, nW);

    moe_gemm_ln<<<dim3(L_ / 32, B_), dim3(256), 0, stream>>>(
        X, Wbf, bias, gamma, beta, ids, out);
}
